// SelfAttention_12635793784876
// MI455X (gfx1250) — matrix operand order repaired and checked by reading, not compiled, not run on hardware
//
#include <hip/hip_runtime.h>
#include <math.h>

typedef _Float16 v16h __attribute__((ext_vector_type(16)));
typedef _Float16 h8   __attribute__((ext_vector_type(8)));
typedef float    v8f  __attribute__((ext_vector_type(8)));
typedef unsigned v8u  __attribute__((ext_vector_type(8)));

// ---------- workspace layout (bytes) ----------
#define OFF_PALL 0u                      // f16 [32768][192]  = 12,582,912
#define OFF_WC   12582912u               // f16 packed Wc frags =    98,304
#define OFF_WO   12681216u               // f16 packed Wo frags =    65,536
#define OFF_GQ   12746752u               // f16 [8][4096][32]  =  2,097,152
#define OFF_FK   14843904u               // f16 [8][1024][32]  =    524,288
#define OFF_HVT  15368192u               // f16 [8][128][1024] =  2,097,152
#define OFF_OWS  17465344u               // f16 [32768][128]   =  8,388,608
// total ~25.9 MB

// Load 16 halfs (two runs of 8) into a v16h fragment register.
__device__ __forceinline__ v16h ld16h2(const _Float16* p0, const _Float16* p1) {
  union { h8 v[2]; v16h w; } t;
  t.v[0] = *(const h8*)p0;
  t.v[1] = *(const h8*)p1;
  return t.w;
}

// Convert two contiguous 8-float runs (p, p+16) into an A fragment.
__device__ __forceinline__ v16h lda_f32(const float* p) {
  v16h a;
#pragma unroll
  for (int i = 0; i < 8; i++) {
    a[i]     = (_Float16)p[i];
    a[8 + i] = (_Float16)p[16 + i];
  }
  return a;
}

__device__ __forceinline__ unsigned pk2h(float a, float b) {
  union { _Float16 h[2]; unsigned u; } t;
  t.h[0] = (_Float16)a; t.h[1] = (_Float16)b;
  return t.u;
}

// ============================================================
// K0: pack weights into B-fragment-ready f16 layout.
// Wc: cols [0:32)=wf, [32:64)=wg, [64:192)=wh  -> [ntile(12)][kk(8)][lane(32)][e(16)]
// Wo: [128][256]                               -> [ntile(16)][kk(4)][lane(32)][e(16)]
// B-fragment element e of lane (hf,m): k = kk*32 + 8*hf + (e&7) + 16*(e>>3), n = ntile*16 + m
// ============================================================
__global__ __launch_bounds__(256) void pack_weights(
    const float* __restrict__ wf, const float* __restrict__ wg,
    const float* __restrict__ wh, const float* __restrict__ wo,
    _Float16* __restrict__ Wc, _Float16* __restrict__ Wo) {
  int idx = blockIdx.x * 256 + threadIdx.x;
  if (idx < 49152) {                       // Wc: 12*8*32*16
    int nt  = idx >> 12;
    int rem = idx & 4095;
    int kk  = rem >> 9;
    int l   = (rem >> 4) & 31;
    int e   = idx & 15;
    int hf  = l >> 4, m = l & 15;
    int k   = kk * 32 + 8 * hf + (e & 7) + 16 * (e >> 3);       // 0..255
    int n   = nt * 16 + m;                 // 0..191
    float v;
    if (n < 32)       v = wf[k * 32 + n];
    else if (n < 64)  v = wg[k * 32 + (n - 32)];
    else              v = wh[k * 128 + (n - 64)];
    Wc[idx] = (_Float16)v;
  } else if (idx < 81920) {                // Wo: 16*4*32*16
    int j   = idx - 49152;
    int nt  = j >> 11;
    int rem = j & 2047;
    int kk  = rem >> 9;
    int l   = (rem >> 4) & 31;
    int e   = j & 15;
    int hf  = l >> 4, m = l & 15;
    int k   = kk * 32 + 8 * hf + (e & 7) + 16 * (e >> 3);       // 0..127
    int n   = nt * 16 + m;                 // 0..255
    Wo[j] = (_Float16)wo[k * 256 + n];
  }
}

// ============================================================
// K1: Pall = f16( x[32768,256] @ Wc[256,192] ).
// 8 waves/block; each wave owns one 16-row M-tile, all 12 N-tiles
// (x read exactly once).  B fragments preloaded into a register
// array so the 24 b128 loads batch in one clause and the 12 WMMAs
// issue back-to-back.  grid = 256 blocks.
// ============================================================
__global__ __launch_bounds__(256) void proj_gemm(
    const float* __restrict__ x, const _Float16* __restrict__ Wc,
    _Float16* __restrict__ Pall) {
  int tid  = threadIdx.x;
  int lane = tid & 31, hf = lane >> 4, m = lane & 15;
  int w    = blockIdx.x * 8 + (tid >> 5);   // wave id = M-tile (0..2047)
  int row  = w * 16 + m;

  v8f acc[12];
  v8f z = {};
#pragma unroll
  for (int n = 0; n < 12; n++) acc[n] = z;

  for (int kk = 0; kk < 8; kk++) {
    const float* pa = x + (size_t)row * 256 + kk * 32 + 8 * hf;
    v16h a = lda_f32(pa);
    v16h bw[12];
#pragma unroll
    for (int nt = 0; nt < 12; nt++) {
      const _Float16* pb = Wc + ((nt * 8 + kk) * 512 + lane * 16);
      bw[nt] = ld16h2(pb, pb + 8);
    }
#pragma unroll
    for (int nt = 0; nt < 12; nt++) {
      acc[nt] = __builtin_amdgcn_wmma_f32_16x16x32_f16(
          false, a, false, bw[nt], (short)0, acc[nt], false, false);
    }
  }
#pragma unroll
  for (int nt = 0; nt < 12; nt++) {
#pragma unroll
    for (int r = 0; r < 8; r++) {
      int orow = w * 16 + 8 * hf + r;
      Pall[(size_t)orow * 192 + nt * 16 + m] = (_Float16)acc[nt][r];
    }
  }
}

// ============================================================
// K2: 2x2 maxpool + repack:
//  fK [b][p][c]   (c<32, cols 0..31 of Pall, pooled)
//  gQ [b][pix][c] (cols 32..63, no pool)
//  hVt[b][c][p]   (cols 64..191, pooled, TRANSPOSED for attn A-frags)
// ============================================================
__global__ __launch_bounds__(256) void pool_pack(
    const _Float16* __restrict__ Pall, _Float16* __restrict__ fK,
    _Float16* __restrict__ gQ, _Float16* __restrict__ hVt) {
  int idx = blockIdx.x * 256 + threadIdx.x;
  if (idx < 262144) {                       // fK : 8*1024*32
    int b   = idx >> 15;
    int rem = idx & 32767;
    int p   = rem >> 5, c = rem & 31;
    int ph  = p >> 5, pw = p & 31;
    size_t r0 = (size_t)(b * 4096 + (ph * 2) * 64 + pw * 2) * 192 + c;
    float v = (float)Pall[r0];
    v = fmaxf(v, (float)Pall[r0 + 192]);
    v = fmaxf(v, (float)Pall[r0 + 64 * 192]);
    v = fmaxf(v, (float)Pall[r0 + 65 * 192]);
    fK[idx] = (_Float16)v;
  } else if (idx < 1310720) {               // gQ : 8*4096*32
    int j   = idx - 262144;
    int b   = j >> 17;
    int rem = j & 131071;
    int pix = rem >> 5, c = rem & 31;
    gQ[j] = Pall[(size_t)(b * 4096 + pix) * 192 + 32 + c];
  } else if (idx < 2359296) {               // hVt : 8*128*1024
    int j   = idx - 1310720;
    int b   = j >> 17;
    int rem = j & 131071;
    int c   = rem >> 10, p = rem & 1023;
    int ph  = p >> 5, pw = p & 31;
    size_t r0 = (size_t)(b * 4096 + (ph * 2) * 64 + pw * 2) * 192 + 64 + c;
    float v = (float)Pall[r0];
    v = fmaxf(v, (float)Pall[r0 + 192]);
    v = fmaxf(v, (float)Pall[r0 + 64 * 192]);
    v = fmaxf(v, (float)Pall[r0 + 65 * 192]);
    hVt[j] = (_Float16)v;
  }
}

// ============================================================
// K3: flash attention (transposed formulation).
// Per wave: 16 queries.  S^T = K(16x32 d) x Q^T(32d x 16q) via WMMA;
// softmax stats per query == per lane; O^T[c][q] += V^T @ P^T.
// V fragments are preloaded right after the score WMMAs so the
// softmax VALU/TRANS work overlaps their load latency; next key
// block is prefetched (global_prefetch_b8).
// grid = 8 batches * 32 q-blocks = 256 blocks, 8 waves each.
// ============================================================
__global__ __launch_bounds__(256) void attn_kernel(
    const _Float16* __restrict__ gQ, const _Float16* __restrict__ fK,
    const _Float16* __restrict__ hVt, _Float16* __restrict__ Ows) {
  int tid  = threadIdx.x;
  int wave = tid >> 5, lane = tid & 31, hf = lane >> 4, m = lane & 15;
  int b    = blockIdx.x >> 5;
  int qblk = blockIdx.x & 31;
  int qbase = (qblk * 8 + wave) * 16;

  // Q as B-fragment: lane holds query qbase+m, d runs 8*hf and 16+8*hf
  const _Float16* gq = gQ + (size_t)(b * 4096 + qbase + m) * 32 + 8 * hf;
  v16h bQ = ld16h2(gq, gq + 16);

  const _Float16* fb = fK + (size_t)b * 1024 * 32;
  const _Float16* vb = hVt + (size_t)b * 128 * 1024;

  float rmax = -INFINITY, rsum = 0.0f;
  v8f z = {};
  v8f acc[8];
#pragma unroll
  for (int j = 0; j < 8; j++) acc[j] = z;

  for (int kb = 0; kb < 1024; kb += 32) {
    // K A-fragments: 16 keys x 32 d each (d runs: 8*hf and 16+8*hf)
    const _Float16* kp0 = fb + (size_t)(kb + m) * 32;
    const _Float16* kp1 = fb + (size_t)(kb + 16 + m) * 32;
    v16h a0 = ld16h2(kp0 + 8 * hf, kp0 + 16 + 8 * hf);
    v16h a1 = ld16h2(kp1 + 8 * hf, kp1 + 16 + 8 * hf);

    v8f s0 = __builtin_amdgcn_wmma_f32_16x16x32_f16(false, a0, false, bQ,
                                                    (short)0, z, false, false);
    v8f s1 = __builtin_amdgcn_wmma_f32_16x16x32_f16(false, a1, false, bQ,
                                                    (short)0, z, false, false);

    // Preload V^T fragments now: their latency overlaps the softmax math.
    v16h aV[8];
#pragma unroll
    for (int ct = 0; ct < 8; ct++) {
      const _Float16* vp = vb + (size_t)(ct * 16 + m) * 1024 + kb;
      aV[ct] = ld16h2(vp + 16 * hf, vp + 16 * hf + 8);
    }
    // Prefetch next block's key rows into cache.
    if (kb + 32 < 1024) {
      __builtin_prefetch(fb + (size_t)(kb + 32 + m) * 32, 0, 0);
      __builtin_prefetch(fb + (size_t)(kb + 48 + m) * 32, 0, 0);
    }

    // online softmax (per query == per lane; combine the two half-groups)
    float bm = -INFINITY;
#pragma unroll
    for (int r = 0; r < 8; r++) bm = fmaxf(bm, fmaxf(s0[r], s1[r]));
    bm = fmaxf(bm, __shfl_xor(bm, 16, 32));
    float mnew  = fmaxf(rmax, bm);
    float scale = __expf(rmax - mnew);
    rsum *= scale;
#pragma unroll
    for (int j = 0; j < 8; j++)
#pragma unroll
      for (int r = 0; r < 8; r++) acc[j][r] *= scale;

    float p0f[8], p1f[8], ls = 0.0f;
#pragma unroll
    for (int r = 0; r < 8; r++) {
      p0f[r] = __expf(s0[r] - mnew);
      p1f[r] = __expf(s1[r] - mnew);
      ls += p0f[r] + p1f[r];
    }
    ls += __shfl_xor(ls, 16, 32);
    rsum += ls;
    rmax = mnew;

    // Build P^T B-fragment (32 keys x 16 queries) with one cross-half shuffle.
    unsigned p0p[4], p1p[4], q0p[4], q1p[4];
#pragma unroll
    for (int i = 0; i < 4; i++) {
      p0p[i] = pk2h(p0f[2 * i], p0f[2 * i + 1]);
      p1p[i] = pk2h(p1f[2 * i], p1f[2 * i + 1]);
    }
#pragma unroll
    for (int i = 0; i < 4; i++) {
      q0p[i] = __shfl_xor(p0p[i], 16, 32);
      q1p[i] = __shfl_xor(p1p[i], 16, 32);
    }
    union { v8u u; v16h h; } bp;
#pragma unroll
    for (int i = 0; i < 4; i++) {
      bp.u[i]     = hf ? q1p[i] : p0p[i];   // keys 16*hf + 0..7
      bp.u[4 + i] = hf ? p1p[i] : q0p[i];   // keys 16*hf + 8..15
    }
    v16h bP = bp.h;

    // O^T accumulation: 8 channel tiles, WMMAs back-to-back.
#pragma unroll
    for (int ct = 0; ct < 8; ct++) {
      acc[ct] = __builtin_amdgcn_wmma_f32_16x16x32_f16(
          false, aV[ct], false, bP, (short)0, acc[ct], false, false);
    }
  }

  // Normalize and store O row-major f16 [32768][128]
  float inv = 1.0f / rsum;
  size_t orow = (size_t)(b * 4096 + qbase + m) * 128;
#pragma unroll
  for (int ct = 0; ct < 8; ct++) {
    unsigned* dst = (unsigned*)(Ows + orow + ct * 16 + 8 * hf);
#pragma unroll
    for (int i = 0; i < 4; i++)
      dst[i] = pk2h(acc[ct][2 * i] * inv, acc[ct][2 * i + 1] * inv);
  }
}

// ============================================================
// K4: out = gamma * (O[32768,128] @ wo[128,256]) + x.
// Each wave: one M-tile, 8 of the 16 N-tiles; B fragments
// preloaded per k-step.  grid = 512 blocks.
// ============================================================
__global__ __launch_bounds__(256) void out_gemm(
    const _Float16* __restrict__ Ows, const _Float16* __restrict__ Wo,
    const float* __restrict__ x, const float* __restrict__ gamma,
    float* __restrict__ out) {
  int tid  = threadIdx.x;
  int lane = tid & 31, hf = lane >> 4, m = lane & 15;
  int w    = blockIdx.x * 8 + (tid >> 5);
  int mtile = w >> 1, grp = w & 1;
  int row  = mtile * 16 + m;

  v8f z = {};
  v8f acc[8];
#pragma unroll
  for (int j = 0; j < 8; j++) acc[j] = z;

  for (int kk = 0; kk < 4; kk++) {
    const _Float16* pa = Ows + (size_t)row * 128 + kk * 32 + 8 * hf;
    v16h a = ld16h2(pa, pa + 16);
    v16h bw[8];
#pragma unroll
    for (int j = 0; j < 8; j++) {
      int nt = grp * 8 + j;
      const _Float16* pb = Wo + ((nt * 4 + kk) * 512 + lane * 16);
      bw[j] = ld16h2(pb, pb + 8);
    }
#pragma unroll
    for (int j = 0; j < 8; j++) {
      acc[j] = __builtin_amdgcn_wmma_f32_16x16x32_f16(
          false, a, false, bw[j], (short)0, acc[j], false, false);
    }
  }

  float gm = gamma[0];
#pragma unroll
  for (int j = 0; j < 8; j++) {
    int nt = grp * 8 + j;
#pragma unroll
    for (int r = 0; r < 8; r++) {
      size_t o = (size_t)(mtile * 16 + 8 * hf + r) * 256 + nt * 16 + m;
      out[o] = gm * acc[j][r] + x[o];
    }
  }
}

// ============================================================
extern "C" void kernel_launch(void* const* d_in, const int* in_sizes, int n_in,
                              void* d_out, int out_size, void* d_ws,
                              size_t ws_size, hipStream_t stream) {
  (void)in_sizes; (void)n_in; (void)out_size; (void)ws_size;
  const float* x     = (const float*)d_in[0];
  const float* wf    = (const float*)d_in[1];
  const float* wg    = (const float*)d_in[2];
  const float* wh    = (const float*)d_in[3];
  const float* wo    = (const float*)d_in[4];
  const float* gamma = (const float*)d_in[5];
  float* out = (float*)d_out;

  char* ws = (char*)d_ws;
  _Float16* Pall = (_Float16*)(ws + OFF_PALL);
  _Float16* Wc   = (_Float16*)(ws + OFF_WC);
  _Float16* Wo   = (_Float16*)(ws + OFF_WO);
  _Float16* gQ   = (_Float16*)(ws + OFF_GQ);
  _Float16* fK   = (_Float16*)(ws + OFF_FK);
  _Float16* hVt  = (_Float16*)(ws + OFF_HVT);
  _Float16* Ows  = (_Float16*)(ws + OFF_OWS);

  pack_weights<<<320, 256, 0, stream>>>(wf, wg, wh, wo, Wc, Wo);
  proj_gemm<<<256, 256, 0, stream>>>(x, Wc, Pall);
  pool_pack<<<9216, 256, 0, stream>>>(Pall, fK, gQ, hVt);
  attn_kernel<<<256, 256, 0, stream>>>(gQ, fK, hVt, Ows);
  out_gemm<<<512, 256, 0, stream>>>(Ows, Wo, x, gamma, out);
}
